// TripletLoss_58420145160899
// MI455X (gfx1250) — compile-verified, matrix operand order checked by reading, not run on hardware
//
#include <hip/hip_runtime.h>
#include <math.h>

typedef __attribute__((ext_vector_type(8))) int v8i;
typedef __attribute__((ext_vector_type(4))) int v4i;   // native clang vector: OK for nontemporal builtin

#define WAVE 32
#define CSPLIT 8  // column splits of the label matrix per row-tile

// ---------------------------------------------------------------------------
// Kernel A: one wave per row. s[row] = sum(f^2) after torch-style normalize:
//   s = sumsq / max(sqrt(sumsq), eps)^2   (== 1 unless the row is ~zero)
// Also zero-initializes counts[] for kernel B's atomics.
// ---------------------------------------------------------------------------
__global__ void rownorm_kernel(const float* __restrict__ feat,
                               float* __restrict__ s,
                               int* __restrict__ counts,
                               int N, int D) {
    const int wave = (blockIdx.x * blockDim.x + threadIdx.x) / WAVE;
    const int lane = threadIdx.x & (WAVE - 1);
    if (wave >= N) return;

    const float4* row = (const float4*)(feat + (size_t)wave * (size_t)D);
    const int nq = D >> 2;
    float acc = 0.0f;
    for (int k = lane; k < nq; k += WAVE) {
        float4 q = row[k];
        acc += q.x * q.x + q.y * q.y + q.z * q.z + q.w * q.w;
    }
    // wave32 butterfly reduction
    for (int off = 16; off >= 1; off >>= 1)
        acc += __shfl_xor(acc, off, WAVE);

    if (lane == 0) {
        float denom = fmaxf(sqrtf(acc), 1e-12f);
        s[wave] = acc / (denom * denom);
        counts[wave] = 0;
    }
}

// ---------------------------------------------------------------------------
// Kernel B: label row-sums via V_WMMA_I32_16X16X64_IU8.
// Each wave owns a 16-row tile and 1/CSPLIT of the columns. Per iteration,
// lane L streams 32 consecutive int32 labels of row (L&15) (offset by 32*(L>>4)
// so the two lane-halves cover disjoint columns), packs them to bytes, and one
// WMMA against an all-ones B accumulates 16x64 labels into the i32 D matrix.
// Row-sum is K-permutation invariant, so the exact A-layout column order is
// irrelevant — only "lane L holds row L%16" matters.
// ---------------------------------------------------------------------------
__global__ void labelsum_wmma_kernel(const int* __restrict__ labels,
                                     int* __restrict__ counts,
                                     int N) {
    const int lane = threadIdx.x & 31;
    const int wave = (int)((blockIdx.x * blockDim.x + threadIdx.x) >> 5);
    const int rowTiles = N >> 4;
    const int rowTile  = wave % rowTiles;
    const int colSplit = wave / rowTiles;
    const int colsPerWave = N / CSPLIT;   // 1024 for N=8192
    const int iters = colsPerWave >> 6;   // 64 columns consumed per WMMA

    const int r = (rowTile << 4) + (lane & 15);
    const int half = lane >> 4;
    const size_t base = (size_t)r * (size_t)N +
                        (size_t)colSplit * (size_t)colsPerWave +
                        (size_t)(half * 32);

    v8i b;
#pragma unroll
    for (int j = 0; j < 8; ++j) b[j] = 0x01010101;  // all-ones u8 B matrix
    v8i c = {};

    for (int it = 0; it < iters; ++it) {
        const v4i* p = (const v4i*)(labels + base + (size_t)it * 64);
        v8i a;
#pragma unroll
        for (int k = 0; k < 8; ++k) {
            v4i q = __builtin_nontemporal_load(&p[k]);   // streaming: TH=NT, b128
            a[k] = (int)((unsigned)q.x | ((unsigned)q.y << 8) |
                         ((unsigned)q.z << 16) | ((unsigned)q.w << 24));
        }
        // 7 args: (sgn_a, A, sgn_b, B, C, reuse_a, reuse_b)
        c = __builtin_amdgcn_wmma_i32_16x16x64_iu8(false, a, false, b, c,
                                                   false, false);
    }

    // D layout (32-bit 16x16): lane 0 VGPR j = D[j, 0]; lane 16 VGPR j = D[8+j, 0].
    if (lane == 0) {
#pragma unroll
        for (int j = 0; j < 8; ++j)
            atomicAdd(&counts[(rowTile << 4) + j], c[j]);
    } else if (lane == 16) {
#pragma unroll
        for (int j = 0; j < 8; ++j)
            atomicAdd(&counts[(rowTile << 4) + 8 + j], c[j]);
    }
}

// ---------------------------------------------------------------------------
// Kernel C: loss_i = relu(s_i * (1 - 2*p_i) + 1), out = mean(loss). One block.
// ---------------------------------------------------------------------------
__global__ void finalize_kernel(const float* __restrict__ s,
                                const int* __restrict__ counts,
                                float* __restrict__ out,
                                int N) {
    __shared__ float red[256];
    const float invN = 1.0f / (float)N;
    float acc = 0.0f;
    for (int i = threadIdx.x; i < N; i += 256) {
        float p = (float)counts[i] * invN;
        float l = fmaf(s[i], 1.0f - 2.0f * p, 1.0f);
        acc += fmaxf(l, 0.0f);
    }
    red[threadIdx.x] = acc;
    __syncthreads();
    for (int off = 128; off >= 1; off >>= 1) {
        if (threadIdx.x < off) red[threadIdx.x] += red[threadIdx.x + off];
        __syncthreads();
    }
    if (threadIdx.x == 0) out[0] = red[0] * invN;
}

// ---------------------------------------------------------------------------
extern "C" void kernel_launch(void* const* d_in, const int* in_sizes, int n_in,
                              void* d_out, int out_size, void* d_ws, size_t ws_size,
                              hipStream_t stream) {
    const float* features = (const float*)d_in[0];
    const int* labels = (const int*)d_in[1];
    float* out = (float*)d_out;

    // in_sizes[1] = N*N (power-of-two N), in_sizes[0] = N*D
    long long nn = (long long)in_sizes[1];
    int N = 1;
    while ((long long)N * (long long)N < nn) N <<= 1;
    int D = in_sizes[0] / N;

    float* s = (float*)d_ws;
    int* counts = (int*)((char*)d_ws + (size_t)N * sizeof(float));

    // A: one wave per row (8 waves / 256-thread block)
    int blocksA = (N * WAVE + 255) / 256;
    rownorm_kernel<<<blocksA, 256, 0, stream>>>(features, s, counts, N, D);

    // B: (N/16) row tiles * CSPLIT column splits waves, 8 waves per block
    int totalWaves = (N >> 4) * CSPLIT;
    int blocksB = (totalWaves + 7) / 8;
    labelsum_wmma_kernel<<<blocksB, 256, 0, stream>>>(labels, counts, N);

    // C: single-block finalize
    finalize_kernel<<<1, 256, 0, stream>>>(s, counts, out, N);
}
